// SingleSelfAttentionHead_40922448396699
// MI455X (gfx1250) — matrix operand order repaired and checked by reading, not compiled, not run on hardware
//
#include <hip/hip_runtime.h>
#include <hip/hip_bf16.h>

// ---------------------------------------------------------------------------
// Single-head causal self-attention, CDNA5 (gfx1250, wave32, WMMA bf16).
//   Kernel 0a: x (f32) -> xb (bf16), streaming convert.
//   Kernel 0b: W{q,k,v} (f32 [D][H]) -> Wt (bf16 [H][D], transposed).
//   Kernel 1 : Q,K = xb @ W (bf16 WMMA, f32 acc), Q pre-scaled by 1/sqrt(H);
//              V stored TRANSPOSED ([b][h][s]) so attention B-fragments are
//              contiguous 32B loads.
//   Kernel 2 : fused causal flash attention, fp32 output.
// All GEMM inner loops are pure b128 loads + v_wmma; conversions happen once.
// ---------------------------------------------------------------------------

typedef __attribute__((ext_vector_type(16))) __bf16          v16bf;
typedef __attribute__((ext_vector_type(16))) unsigned short  v16us;
typedef __attribute__((ext_vector_type(8), aligned(16))) unsigned short v8us;
typedef __attribute__((ext_vector_type(4), aligned(8)))  unsigned short v4us;
typedef __attribute__((ext_vector_type(8)))  float           v8f;
typedef __attribute__((ext_vector_type(4)))  float           fvec4;

#define SEQ    4096
#define DMODEL 1024
#define HEAD   64
#define NB     4
#define NROWS  (NB * SEQ)   // B * S = 16384

// round-to-nearest-even f32 -> bf16 (used only in the convert/epilogue paths)
__device__ __forceinline__ unsigned short f2bf(float f) {
  unsigned u = __builtin_bit_cast(unsigned, f);
  u += 0x7FFFu + ((u >> 16) & 1u);
  return (unsigned short)(u >> 16);
}

__device__ __forceinline__ v8f wmma_bf16(v16us a, v16us b, v8f c) {
  return __builtin_amdgcn_wmma_f32_16x16x32_bf16(
      /*neg_a=*/false, __builtin_bit_cast(v16bf, a),
      /*neg_b=*/false, __builtin_bit_cast(v16bf, b),
      /*c_mod=*/(short)0, c, /*reuse_a=*/false, /*reuse_b=*/false);
}

// A-layout fragment: 16B at p, 16B at p+8 elements (32B contiguous)
__device__ __forceinline__ v16us ld_frag_gap(const unsigned short* p) {
  v16us f;
  *(v8us*)&f       = *(const v8us*)p;
  *((v8us*)&f + 1) = *(const v8us*)(p + 8);
  return f;
}
// B-layout fragment: 32B contiguous
__device__ __forceinline__ v16us ld_frag32(const unsigned short* p) {
  v16us f;
  *(v8us*)&f       = *(const v8us*)p;
  *((v8us*)&f + 1) = *(const v8us*)(p + 8);
  return f;
}

// ---------------------------------------------------------------------------
// Kernel 0a: x f32 -> bf16, 8 elements/thread, b128 in / b128 out.
// ---------------------------------------------------------------------------
__global__ __launch_bounds__(256) void cvt_x_kernel(
    const float* __restrict__ x, unsigned short* __restrict__ xb) {
  const long i = ((long)blockIdx.x * 256 + threadIdx.x) * 8;
  fvec4 a = *(const fvec4*)(x + i);
  fvec4 b = *(const fvec4*)(x + i + 4);
  v8us t;
#pragma unroll
  for (int j = 0; j < 4; ++j) { t[j] = f2bf(a[j]); t[4 + j] = f2bf(b[j]); }
  *(v8us*)(xb + i) = t;
}

// ---------------------------------------------------------------------------
// Kernel 0b: W [D][H] f32 -> Wt [H][D] bf16 (3 matrices concatenated).
// One block per (matrix, output row h).
// ---------------------------------------------------------------------------
__global__ __launch_bounds__(256) void cvt_w_kernel(
    const float* __restrict__ Wq, const float* __restrict__ Wk,
    const float* __restrict__ Wv, unsigned short* __restrict__ Wt) {
  const int m = blockIdx.x >> 6;        // matrix 0..2
  const int h = blockIdx.x & 63;        // output row (0..H-1)
  const float* W = (m == 0) ? Wq : (m == 1) ? Wk : Wv;
  unsigned short* dst = Wt + ((long)m * HEAD + h) * DMODEL;
  const int k0 = threadIdx.x * 4;       // 256 threads * 4 = D
  v4us t;
#pragma unroll
  for (int i = 0; i < 4; ++i) t[i] = f2bf(W[(long)(k0 + i) * HEAD + h]);
  *(v4us*)(dst + k0) = t;
}

// ---------------------------------------------------------------------------
// Kernel 1: QKV projection. One block = 16 rows; wave w owns h-tile w.
// Inner loop: 8 x b128 loads + 3 x v_wmma, no per-element VALU.
// ---------------------------------------------------------------------------
__global__ __launch_bounds__(128) void qkv_proj_kernel(
    const unsigned short* __restrict__ xb, const unsigned short* __restrict__ Wt,
    unsigned short* __restrict__ Qb, unsigned short* __restrict__ Kb,
    unsigned short* __restrict__ Vt) {
  const int lane = threadIdx.x & 31;
  const int w    = threadIdx.x >> 5;   // h-tile 0..3
  const int half = lane >> 4;
  const int n16  = lane & 15;
  const long R0  = (long)blockIdx.x * 16;
  const int  c   = w * 16 + n16;       // output column (N of B/C fragment)

  const unsigned short* Wqt = Wt + (long)c * DMODEL;                    // row c
  const unsigned short* Wkt = Wt + ((long)HEAD + c) * DMODEL;
  const unsigned short* Wvt = Wt + (2L * HEAD + c) * DMODEL;
  const unsigned short* xrow = xb + (R0 + n16) * (long)DMODEL;          // A: M = lane&15

  v8f q = {}, k = {}, v = {};
  for (int k0 = 0; k0 < DMODEL; k0 += 32) {
    const int ka = k0 + (half ? 16 : 0);   // A-layout half offset
    const int kb = k0 + (half ? 16 : 0);   // B-layout half offset
    v16us a  = ld_frag_gap(xrow + ka);
    v16us bq = ld_frag32(Wqt + kb);
    v16us bk = ld_frag32(Wkt + kb);
    v16us bv = ld_frag32(Wvt + kb);
    q = wmma_bf16(a, bq, q);
    k = wmma_bf16(a, bk, k);
    v = wmma_bf16(a, bv, v);
  }

  const long bidx = R0 / SEQ;              // batch (tiles never straddle)
  const long srow0 = R0 - bidx * SEQ;
  unsigned short* vtrow = Vt + (bidx * HEAD + c) * (long)SEQ;
#pragma unroll
  for (int r = 0; r < 8; ++r) {            // C layout: M = r + 8*half, N = c
    const long row = R0 + r + 8 * half;
    Qb[row * HEAD + c] = f2bf(q[r] * 0.125f);   // fold 1/sqrt(64) into Q
    Kb[row * HEAD + c] = f2bf(k[r]);
    vtrow[srow0 + r + 8 * half] = f2bf(v[r]);   // transposed V: [b][h][s]
  }
}

// ---------------------------------------------------------------------------
// Kernel 2: fused causal flash attention.
// Block = (batch b, 64 query rows); wave w owns 16 query rows, full H=64.
// ---------------------------------------------------------------------------
__global__ __launch_bounds__(128) void flash_attn_kernel(
    const unsigned short* __restrict__ Qb, const unsigned short* __restrict__ Kb,
    const unsigned short* __restrict__ Vt, float* __restrict__ out) {
  __shared__ unsigned short Psh[4 * 16 * 64];   // per-wave 16x64 bf16 P tile

  const int lane = threadIdx.x & 31;
  const int w    = threadIdx.x >> 5;
  const int half = lane >> 4;
  const int n16  = lane & 15;

  const int  qb      = blockIdx.x & 63;        // S/64 blocks per batch
  const int  b       = blockIdx.x >> 6;
  const long rowBase = (long)b * SEQ;
  const int  rq      = qb * 64 + w * 16;       // first query row of this wave

  // Q in A-layout, resident for the whole kernel (2 k-steps over h)
  v16us qf[2];
#pragma unroll
  for (int kk = 0; kk < 2; ++kk)
    qf[kk] = ld_frag_gap(Qb + (rowBase + rq + n16) * (long)HEAD +
                         32 * kk + (half ? 16 : 0));

  float m8[8], l8[8];
  v8f o[4];
#pragma unroll
  for (int r = 0; r < 8; ++r) { m8[r] = -1e30f; l8[r] = 0.f; }
#pragma unroll
  for (int n = 0; n < 4; ++n) o[n] = {};

  unsigned short* Pw = &Psh[w * 16 * 64];

  for (int kb = 0; kb <= qb; ++kb) {
    const int kbase = kb * 64;
    float s[4][8];

    // ---- S = Q * K^T (B fragment: N = key, K-dim = h, contiguous in mem)
#pragma unroll
    for (int j2 = 0; j2 < 4; ++j2) {
      v8f acc = {};
#pragma unroll
      for (int kk = 0; kk < 2; ++kk) {
        v16us kf = ld_frag32(Kb + (rowBase + kbase + j2 * 16 + n16) * (long)HEAD +
                             32 * kk + (half ? 16 : 0));
        acc = wmma_bf16(qf[kk], kf, acc);
      }
      const int keyg = kbase + j2 * 16 + n16;
#pragma unroll
      for (int r = 0; r < 8; ++r) {
        float sv = acc[r];
        if (kb == qb && keyg > rq + r + 8 * half) sv = -1e30f;  // causal mask
        s[j2][r] = sv;
      }
    }

    // ---- online softmax (row = r + 8*half lives in one 16-lane half)
    float escale[8];
#pragma unroll
    for (int r = 0; r < 8; ++r) {
      float mx = fmaxf(fmaxf(s[0][r], s[1][r]), fmaxf(s[2][r], s[3][r]));
#pragma unroll
      for (int d = 1; d < 16; d <<= 1) mx = fmaxf(mx, __shfl_xor(mx, d, 32));
      const float mn = fmaxf(m8[r], mx);
      escale[r] = __expf(m8[r] - mn);
      m8[r] = mn;
    }
#pragma unroll
    for (int j2 = 0; j2 < 4; ++j2)
#pragma unroll
      for (int r = 0; r < 8; ++r) s[j2][r] = __expf(s[j2][r] - m8[r]);
#pragma unroll
    for (int r = 0; r < 8; ++r) {
      float rs = (s[0][r] + s[1][r]) + (s[2][r] + s[3][r]);
#pragma unroll
      for (int d = 1; d < 16; d <<= 1) rs += __shfl_xor(rs, d, 32);
      l8[r] = l8[r] * escale[r] + rs;
    }
#pragma unroll
    for (int n = 0; n < 4; ++n)
#pragma unroll
      for (int r = 0; r < 8; ++r) o[n][r] *= escale[r];

    // ---- P: C-layout f32 -> LDS bf16 (row-major 16x64, wave-private)
#pragma unroll
    for (int j2 = 0; j2 < 4; ++j2)
#pragma unroll
      for (int r = 0; r < 8; ++r)
        Pw[(r + 8 * half) * 64 + j2 * 16 + n16] = f2bf(s[j2][r]);

    asm volatile("" ::: "memory");  // order punned LDS store->load pair

    // reload P in A-layout (same-wave LDS ops are kept in order by HW)
    v16us pf[2];
#pragma unroll
    for (int kk = 0; kk < 2; ++kk)
      pf[kk] = ld_frag_gap(Pw + n16 * 64 + 32 * kk + (half ? 16 : 0));

    // ---- O += P * V  (V is pre-transposed: fragment = 32B contiguous)
#pragma unroll
    for (int kk = 0; kk < 2; ++kk) {
      const int keyoff = kbase + 32 * kk + (half ? 16 : 0);
#pragma unroll
      for (int n = 0; n < 4; ++n) {
        v16us vf = ld_frag32(Vt + ((long)b * HEAD + n * 16 + n16) * SEQ + keyoff);
        o[n] = wmma_bf16(pf[kk], vf, o[n]);
      }
    }

    if (kb < qb)  // pull next K block toward the WGP while WMMA drains
      __builtin_prefetch(Kb + (rowBase + kbase + 64 + n16 * 4) * (long)HEAD, 0, 1);
  }

  // ---- epilogue: normalize and store fp32
#pragma unroll
  for (int r = 0; r < 8; ++r) {
    const float inv = 1.f / l8[r];
    const long row = rowBase + rq + r + 8 * half;
#pragma unroll
    for (int n = 0; n < 4; ++n)
      out[row * (long)HEAD + n * 16 + n16] = o[n][r] * inv;
  }
}

// ---------------------------------------------------------------------------
extern "C" void kernel_launch(void* const* d_in, const int* in_sizes, int n_in,
                              void* d_out, int out_size, void* d_ws, size_t ws_size,
                              hipStream_t stream) {
  const float* x  = (const float*)d_in[0];
  const float* Wq = (const float*)d_in[1];
  const float* Wk = (const float*)d_in[2];
  const float* Wv = (const float*)d_in[3];
  float* out = (float*)d_out;

  // workspace layout (ushort units): xb | Wt | Qb | Kb | Vt  (~38.5 MB)
  unsigned short* xb = (unsigned short*)d_ws;
  unsigned short* Wt = xb + (size_t)NROWS * DMODEL;
  unsigned short* Qb = Wt + (size_t)3 * HEAD * DMODEL;
  unsigned short* Kb = Qb + (size_t)NROWS * HEAD;
  unsigned short* Vt = Kb + (size_t)NROWS * HEAD;

  cvt_x_kernel<<<(NROWS * DMODEL) / (256 * 8), 256, 0, stream>>>(x, xb);
  cvt_w_kernel<<<3 * HEAD, 256, 0, stream>>>(Wq, Wk, Wv, Wt);
  qkv_proj_kernel<<<NROWS / 16, 128, 0, stream>>>(xb, Wt, Qb, Kb, Vt);
  flash_attn_kernel<<<NB * (SEQ / 64), 128, 0, stream>>>(Qb, Kb, Vt, out);
}
